// Net_DT_39109972197864
// MI455X (gfx1250) — compile-verified, matrix operand order checked by reading, not run on hardware
//
#include <hip/hip_runtime.h>

// ---------------- types ----------------
typedef __attribute__((ext_vector_type(16))) __bf16 v16bf;
typedef __attribute__((ext_vector_type(8)))  __bf16 v8bf;
typedef __attribute__((ext_vector_type(4)))  __bf16 v4bf;
typedef __attribute__((ext_vector_type(8)))  float  v8f;
typedef __attribute__((ext_vector_type(4)))  float  v4f;

#define B_DIM 8192
#define D_DIM 2048
#define O_DIM 1000
#define NPAD  1024      // padded inner-node / leaf / output-col count
#define DEPTHL 10

// GEMM tiling
#define BM 128
#define BN 128
#define BK 32
#define LDSS 48         // bf16 elements per LDS row (32 + 16 pad); row = 96B, 16B aligned

// ---- CDNA5 async global->LDS copy (ASYNCcnt-tracked, no VGPR data staging) ----
__device__ __forceinline__ void async_copy_b128(unsigned int lds_byte_off,
                                                const void* gptr) {
    asm volatile("global_load_async_to_lds_b128 %0, %1, off"
                 :: "v"(lds_byte_off), "v"((unsigned long long)(uintptr_t)gptr)
                 : "memory");
}
__device__ __forceinline__ void wait_async_le4() {
    asm volatile("s_wait_asynccnt 0x4" ::: "memory");
}
__device__ __forceinline__ void wait_async_0() {
    asm volatile("s_wait_asynccnt 0x0" ::: "memory");
}
// low 32 bits of a generic pointer to LDS == hardware LDS byte offset
__device__ __forceinline__ unsigned int lds_offset(const void* p) {
    return (unsigned int)(uintptr_t)p;
}

// ---------------- conversion kernels ----------------
__global__ __launch_bounds__(256) void cvt_x_kernel(const float* __restrict__ X,
                                                    __bf16* __restrict__ Xb, int n4) {
    int i = blockIdx.x * blockDim.x + threadIdx.x;
    int stride = gridDim.x * blockDim.x;
    for (; i < n4; i += stride) {
        v4f x = *(const v4f*)(X + (size_t)i * 4);
        v4bf y;
        y.x = (__bf16)x.x; y.y = (__bf16)x.y; y.z = (__bf16)x.z; y.w = (__bf16)x.w;
        *(v4bf*)(Xb + (size_t)i * 4) = y;
    }
}

// W_inner: (1023, 2049) -> Wb: (1024, 2048) bf16 (col 0 split off as bias), pad row 1023 with 0
__global__ __launch_bounds__(256) void cvt_wi_kernel(const float* __restrict__ Wi,
                                                     __bf16* __restrict__ Wb,
                                                     float* __restrict__ bias) {
    int row = blockIdx.x;            // 0..1023
    bool valid = row < 1023;
    for (int c = threadIdx.x; c < D_DIM; c += 256)
        Wb[(size_t)row * D_DIM + c] = valid ? (__bf16)Wi[(size_t)row * (D_DIM + 1) + 1 + c]
                                            : (__bf16)0.0f;
    if (threadIdx.x == 0) bias[row] = valid ? Wi[(size_t)row * (D_DIM + 1)] : 0.0f;
}

// W_leaf: (1000, 1024) -> (1024, 1024) bf16, pad rows >= 1000 with 0
__global__ __launch_bounds__(256) void cvt_wl_kernel(const float* __restrict__ Wl,
                                                     __bf16* __restrict__ Wlb) {
    int row = blockIdx.x;            // 0..1023
    bool valid = row < O_DIM;
    for (int c = threadIdx.x; c < NPAD; c += 256)
        Wlb[(size_t)row * NPAD + c] = valid ? (__bf16)Wl[(size_t)row * NPAD + c] : (__bf16)0.0f;
}

__global__ __launch_bounds__(256) void zero_sums_kernel(float* __restrict__ gnum,
                                                        float* __restrict__ gden) {
    int i = blockIdx.x * 256 + threadIdx.x;
    if (i < NPAD) { gnum[i] = 0.0f; gden[i] = 0.0f; }
}

// ---------------- WMMA GEMM: C(MxN) = A(MxK) * Bm(NxK)^T ----------------
// Double-buffered LDS; tiles staged with global_load_async_to_lds_b128 (ASYNCcnt).
// MODE 1: epilogue p = sigmoid(acc + bias[n]), store f32 (stride ldc)
// MODE 0: plain f32 store, masked to n < nlim
template <int MODE>
__global__ __launch_bounds__(256) void gemm_wmma_kernel(
        const __bf16* __restrict__ A, const __bf16* __restrict__ Bm,
        const float* __restrict__ bias, float* __restrict__ C,
        int K, int ldc, int nlim) {
    __shared__ __bf16 sA[2][BM * LDSS];
    __shared__ __bf16 sB[2][BN * LDSS];

    const int tid   = threadIdx.x;
    const int lane  = tid & 31;
    const int wave  = tid >> 5;        // 0..7
    const int waveM = wave & 3;        // 4 waves along M
    const int waveN = wave >> 2;       // 2 waves along N
    const int bm = blockIdx.x * BM;
    const int bn = blockIdx.y * BN;

    v8f acc[2][4];
    #pragma unroll
    for (int tm = 0; tm < 2; ++tm)
        #pragma unroll
        for (int tn = 0; tn < 4; ++tn) {
            v8f z = {0.f, 0.f, 0.f, 0.f, 0.f, 0.f, 0.f, 0.f};
            acc[tm][tn] = z;
        }

    // Per-thread chunk geometry: 512 chunks of 8 bf16 (16 B) cover a 128x32 tile.
    // Thread t handles chunks t and t+256 for both A and B.
    const int r0c  = tid >> 2;              // chunk tid     -> row
    const int r1c  = (tid + 256) >> 2;      // chunk tid+256 -> row
    const int colb = (tid & 3) * 8;         // bf16 column within row (same for both)
    const unsigned int sA_base = lds_offset(&sA[0][0]);
    const unsigned int sB_base = lds_offset(&sB[0][0]);
    const unsigned int bufA_sz = BM * LDSS * 2;  // bytes per A buffer
    const unsigned int bufB_sz = BN * LDSS * 2;

    auto issue_tile = [&](int kt, int buf) {
        const int k0 = kt * BK;
        const __bf16* gA0 = A  + (size_t)(bm + r0c) * K + k0 + colb;
        const __bf16* gA1 = A  + (size_t)(bm + r1c) * K + k0 + colb;
        const __bf16* gB0 = Bm + (size_t)(bn + r0c) * K + k0 + colb;
        const __bf16* gB1 = Bm + (size_t)(bn + r1c) * K + k0 + colb;
        const unsigned int la0 = sA_base + buf * bufA_sz + r0c * (LDSS * 2) + colb * 2;
        const unsigned int la1 = sA_base + buf * bufA_sz + r1c * (LDSS * 2) + colb * 2;
        const unsigned int lb0 = sB_base + buf * bufB_sz + r0c * (LDSS * 2) + colb * 2;
        const unsigned int lb1 = sB_base + buf * bufB_sz + r1c * (LDSS * 2) + colb * 2;
        async_copy_b128(la0, gA0);
        async_copy_b128(la1, gA1);
        async_copy_b128(lb0, gB0);
        async_copy_b128(lb1, gB1);
    };

    // Fragment layout (16-bit A/B, wave32): lane<16 -> K {0..7,16..23}, lane>=16 -> K {8..15,24..31}
    const int hof  = (lane >> 4) * 8;
    const int lrow = lane & 15;
    auto loadA = [&](int buf, int tm) -> v16bf {
        const __bf16* base = &sA[buf][(waveM * 32 + tm * 16 + lrow) * LDSS];
        v8bf lo = *(const v8bf*)(base + hof);
        v8bf hi = *(const v8bf*)(base + 16 + hof);
        return __builtin_shufflevector(lo, hi, 0, 1, 2, 3, 4, 5, 6, 7,
                                               8, 9, 10, 11, 12, 13, 14, 15);
    };
    auto loadB = [&](int buf, int tn) -> v16bf {
        const __bf16* base = &sB[buf][(waveN * 64 + tn * 16 + lrow) * LDSS];
        v8bf lo = *(const v8bf*)(base + hof);
        v8bf hi = *(const v8bf*)(base + 16 + hof);
        return __builtin_shufflevector(lo, hi, 0, 1, 2, 3, 4, 5, 6, 7,
                                               8, 9, 10, 11, 12, 13, 14, 15);
    };

    const int KT = K / BK;
    issue_tile(0, 0);

    for (int kt = 0; kt < KT; ++kt) {
        const int cur = kt & 1;
        if (kt + 1 < KT) {
            issue_tile(kt + 1, cur ^ 1);   // prefetch next tile into other buffer
            wait_async_le4();              // my 4 copies for tile kt have landed
        } else {
            wait_async_0();
        }
        __syncthreads();                   // everyone's tile-kt copies landed

        v16bf af[2], bfr[4];
        #pragma unroll
        for (int tm = 0; tm < 2; ++tm) af[tm] = loadA(cur, tm);
        #pragma unroll
        for (int tn = 0; tn < 4; ++tn) bfr[tn] = loadB(cur, tn);

        #pragma unroll
        for (int tm = 0; tm < 2; ++tm)
            #pragma unroll
            for (int tn = 0; tn < 4; ++tn)
                acc[tm][tn] = __builtin_amdgcn_wmma_f32_16x16x32_bf16(
                    false, af[tm], false, bfr[tn], (short)0, acc[tm][tn], false, false);

        __syncthreads();                   // reads of buf cur done before it is re-filled
    }

    // epilogue: C/D layout: VGPR g -> M = g + 8*(lane>>4), N = lane&15
    const int mofs = 8 * (lane >> 4);
    #pragma unroll
    for (int tm = 0; tm < 2; ++tm) {
        #pragma unroll
        for (int tn = 0; tn < 4; ++tn) {
            int ncol = bn + waveN * 64 + tn * 16 + lrow;
            int mrow = bm + waveM * 32 + tm * 16 + mofs;
            #pragma unroll
            for (int g = 0; g < 8; ++g) {
                float v = acc[tm][tn][g];
                if (MODE == 1) {
                    v = 1.0f / (1.0f + __expf(-(v + bias[ncol])));
                    C[(size_t)(mrow + g) * ldc + ncol] = v;
                } else {
                    if (ncol < nlim) C[(size_t)(mrow + g) * ldc + ncol] = v;
                }
            }
        }
    }
}

// ---------------- tree recursion + per-node batch sums ----------------
#define TROWS 4
__global__ __launch_bounds__(256) void tree_kernel(const float* __restrict__ P,
                                                   __bf16* __restrict__ Mu,
                                                   float* __restrict__ gnum,
                                                   float* __restrict__ gden) {
    __shared__ float mu[2][TROWS][NPAD];
    __shared__ float snum[NPAD];
    __shared__ float sden[NPAD];

    const int tid = threadIdx.x;
    const int r0  = blockIdx.x * TROWS;

    for (int i = tid; i < NPAD; i += 256) { snum[i] = 0.0f; sden[i] = 0.0f; }
    if (tid < TROWS) mu[0][tid][0] = 1.0f;
    __syncthreads();

    int cur = 0;
    for (int layer = 0; layer < DEPTHL; ++layer) {
        const int n = 1 << layer;
        const int begin = n - 1;
        for (int idx = tid; idx < TROWS * n; idx += 256) {
            int r = idx >> layer;
            int i = idx & (n - 1);
            float m = mu[cur][r][i];
            float p = P[(size_t)(r0 + r) * NPAD + begin + i];
            atomicAdd(&snum[begin + i], p * m);   // ds_add_f32
            atomicAdd(&sden[begin + i], m);
            mu[cur ^ 1][r][2 * i]     = m * p;
            mu[cur ^ 1][r][2 * i + 1] = m * (1.0f - p);
        }
        __syncthreads();
        cur ^= 1;
    }

    for (int idx = tid; idx < TROWS * NPAD; idx += 256) {
        int r = idx >> 10;
        int j = idx & (NPAD - 1);
        Mu[(size_t)(r0 + r) * NPAD + j] = (__bf16)mu[cur][r][j];
    }
    for (int i = tid; i < NPAD - 1; i += 256) {
        atomicAdd(&gnum[i], snum[i]);
        atomicAdd(&gden[i], sden[i]);
    }
}

// ---------------- penalty reduction ----------------
__global__ __launch_bounds__(256) void penalty_kernel(const float* __restrict__ gnum,
                                                      const float* __restrict__ gden,
                                                      float* __restrict__ out) {
    __shared__ float red[256];
    float acc = 0.0f;
    for (int node = threadIdx.x; node < NPAD - 1; node += 256) {
        int layer = 31 - __clz(node + 1);
        float coeff = 0.001f * exp2f(-(float)layer);
        float den = gden[node];
        float a0 = gnum[node] / den;      // alpha for p branch
        float a1 = 1.0f - a0;             // num0 + num1 == den
        acc -= 0.5f * coeff * (__logf(a0) + log1pf(-a0) + __logf(a1) + log1pf(-a1));
    }
    red[threadIdx.x] = acc;
    __syncthreads();
    for (int s = 128; s > 0; s >>= 1) {
        if (threadIdx.x < s) red[threadIdx.x] += red[threadIdx.x + s];
        __syncthreads();
    }
    if (threadIdx.x == 0) out[0] = red[0];
}

// ---------------- launch ----------------
extern "C" void kernel_launch(void* const* d_in, const int* in_sizes, int n_in,
                              void* d_out, int out_size, void* d_ws, size_t ws_size,
                              hipStream_t stream) {
    const float* X  = (const float*)d_in[0];   // (8192, 2048)
    const float* Wi = (const float*)d_in[1];   // (1023, 2049)
    const float* Wl = (const float*)d_in[2];   // (1000, 1024)
    float* out = (float*)d_out;                // 8192*1000 y_pred + 1 penalty

    char* ws = (char*)d_ws;
    size_t off = 0;
    auto carve = [&](size_t bytes) {
        void* p = ws + off;
        off += (bytes + 255) & ~(size_t)255;
        return p;
    };
    __bf16* Xb   = (__bf16*)carve((size_t)B_DIM * D_DIM * 2);   // 32 MB
    __bf16* Wb   = (__bf16*)carve((size_t)NPAD * D_DIM * 2);    // 4 MB
    float*  bias = (float*) carve((size_t)NPAD * 4);
    float*  P    = (float*) carve((size_t)B_DIM * NPAD * 4);    // 32 MB
    __bf16* Mu   = (__bf16*)carve((size_t)B_DIM * NPAD * 2);    // 16 MB
    __bf16* Wlb  = (__bf16*)carve((size_t)NPAD * NPAD * 2);     // 2 MB
    float*  gnum = (float*) carve((size_t)NPAD * 4);
    float*  gden = (float*) carve((size_t)NPAD * 4);

    cvt_x_kernel<<<4096, 256, 0, stream>>>(X, Xb, (B_DIM * D_DIM) / 4);
    cvt_wi_kernel<<<NPAD, 256, 0, stream>>>(Wi, Wb, bias);
    cvt_wl_kernel<<<NPAD, 256, 0, stream>>>(Wl, Wlb);
    zero_sums_kernel<<<(NPAD + 255) / 256, 256, 0, stream>>>(gnum, gden);

    // GEMM1: P = sigmoid(Xb @ Wb^T + bias), M=8192 N=1024 K=2048
    dim3 g1(B_DIM / BM, NPAD / BN);
    gemm_wmma_kernel<1><<<g1, 256, 0, stream>>>(Xb, Wb, bias, P, D_DIM, NPAD, NPAD);

    // tree: mu leaves + per-node sums
    tree_kernel<<<B_DIM / TROWS, 256, 0, stream>>>(P, Mu, gnum, gden);

    // penalty scalar -> d_out[8192000]
    penalty_kernel<<<1, 256, 0, stream>>>(gnum, gden, out + (size_t)B_DIM * O_DIM);

    // GEMM2: y = Mu @ Wlb^T, M=8192 N=1024(masked to 1000) K=1024
    dim3 g2(B_DIM / BM, NPAD / BN);
    gemm_wmma_kernel<0><<<g2, 256, 0, stream>>>(Mu, Wlb, nullptr, out, NPAD, O_DIM, O_DIM);
}
